// OCDLosses_52982716564244
// MI455X (gfx1250) — compile-verified, matrix operand order checked by reading, not run on hardware
//
#include <hip/hip_runtime.h>

#define S_DIM 64
#define B_DIM 128
#define L_DIM 8192
#define TEMP_ 10.0f
#define EPS_ 1e-8f

typedef __attribute__((ext_vector_type(2)))  float    v2f;
typedef __attribute__((ext_vector_type(8)))  float    v8f;
typedef __attribute__((ext_vector_type(16))) _Float16 v16h;

#if defined(__has_builtin)
#if __has_builtin(__builtin_amdgcn_wmma_f32_16x16x4_f32)
#define OCD_WMMA_F32X4 1
#endif
#endif

// ---------------------------------------------------------------------------
// Kernel 1: per-batch count of targets (nTgt[b] = sum_l targets[b,l])
// ---------------------------------------------------------------------------
__global__ __launch_bounds__(256) void ocd_tgt_count(const int* __restrict__ targets,
                                                     int* __restrict__ nTgt) {
    const int b = blockIdx.x;
    int sum = 0;
    for (int l = threadIdx.x; l < L_DIM; l += 256)
        sum += targets[(size_t)b * L_DIM + l];
    __shared__ int sh[256];
    sh[threadIdx.x] = sum;
    __syncthreads();
    for (int st = 128; st; st >>= 1) {
        if (threadIdx.x < st) sh[threadIdx.x] += sh[threadIdx.x + st];
        __syncthreads();
    }
    if (threadIdx.x == 0) nTgt[b] = sh[0];
}

// ---------------------------------------------------------------------------
// Kernel 2: register-resident scan over S per batch element.
// Records, for each q-index s: cum_shift cs, exact q[s,b,0], active count
// (targets=1 & not consumed, excluding l=0), mask; plus the consumed-column
// list (<=63 entries) with consumption step for the gather correction.
// ---------------------------------------------------------------------------
__global__ __launch_bounds__(128) void ocd_scan(const int* __restrict__ symbols,
                                                const int* __restrict__ targets,
                                                const int* __restrict__ nTgt,
                                                float* __restrict__ cs_o,
                                                float* __restrict__ q0_o,
                                                float* __restrict__ nA_o,
                                                float* __restrict__ mask_o,
                                                int* __restrict__ cCols,
                                                int* __restrict__ cSteps,
                                                int* __restrict__ cCnt) {
    const int b = threadIdx.x;
    if (b >= B_DIM) return;
    int cols[S_DIM];
    int steps[S_DIM];
    int cnt = 0;
    int rem = nTgt[b];
    const int tgt0 = targets[(size_t)b * L_DIM];
    int consumed0 = 0;
    float cs = 0.f;
    float q0v = (float)tgt0 - 1.0f;
    float m = 1.f;
    for (int i = 0; i < S_DIM; ++i) {
        const int idx = i * B_DIM + b;
        cs_o[idx]   = cs;
        q0_o[idx]   = q0v;
        nA_o[idx]   = (float)(rem - ((tgt0 && !consumed0) ? 1 : 0));
        mask_o[idx] = m;
        if (i == S_DIM - 1) break;
        const int sym = symbols[i * B_DIM + b];
        int c = 0;
        if (targets[(size_t)b * L_DIM + sym] != 0) {
            bool dup = false;
            for (int j = 0; j < cnt; ++j)
                if (cols[j] == sym) { dup = true; break; }
            if (!dup) {
                c = 1;
                cols[cnt] = sym;
                steps[cnt] = i + 1;  // affects q-indices >= i+1
                ++cnt;
                --rem;
                if (sym == 0) consumed0 = 1;
            }
        }
        const float isEnd = (rem == 0) ? 1.f : 0.f;
        // q[0] update: uniform (c-1), minus is_correct at l==sym, plus is_end
        q0v += (float)(c - 1) - ((sym == 0) ? (float)c : 0.f) + isEnd;
        cs  += (float)(c - 1);
        if (sym == 0) m = 0.f;
    }
    cCnt[b] = cnt;
    for (int j = 0; j < cnt; ++j) {
        cCols[b * S_DIM + j]  = cols[j];
        cSteps[b * S_DIM + j] = steps[j];
    }
}

// ---------------------------------------------------------------------------
// Kernel 3: streaming row reductions via WMMA.
// One block (4 waves) per (s,b) row. Each wave sums 2048 contiguous f32
// through V_WMMA with A = all-ones: sum(D) == 16 * sum(B-tile).
// Produces sAll[row] = sum_l outputs, sTgt[row] = sum_l targets*outputs.
// ---------------------------------------------------------------------------
__global__ __launch_bounds__(128) void ocd_row_reduce(const float* __restrict__ outputs,
                                                      const int* __restrict__ targets,
                                                      float* __restrict__ sAll,
                                                      float* __restrict__ sTgt) {
    const int row  = blockIdx.x;          // row = s*B + b
    const int b    = row % B_DIM;
    const int lane = threadIdx.x & 31;
    const int wave = threadIdx.x >> 5;
    const float* __restrict__ rowO = outputs + (size_t)row * L_DIM;
    const int*   __restrict__ rowT = targets + (size_t)b * L_DIM;
    const int base = wave * (L_DIM / 4);  // 2048 floats per wave

    v8f dA = {};
    v8f dT = {};
#if OCD_WMMA_F32X4
    const v2f ones2 = {1.0f, 1.0f};
    for (int it = 0; it < 16; ++it) {
        const int idx = base + it * 128 + lane * 4;
        const float4 o = *reinterpret_cast<const float4*>(rowO + idx);
        const int4   t = *reinterpret_cast<const int4*>(rowT + idx);
        v2f b0  = {o.x, o.y};
        v2f b1  = {o.z, o.w};
        v2f tb0 = {o.x * (float)t.x, o.y * (float)t.y};
        v2f tb1 = {o.z * (float)t.z, o.w * (float)t.w};
        dA = __builtin_amdgcn_wmma_f32_16x16x4_f32(false, ones2, false, b0,  (short)0, dA, false, false);
        dT = __builtin_amdgcn_wmma_f32_16x16x4_f32(false, ones2, false, tb0, (short)0, dT, false, false);
        dA = __builtin_amdgcn_wmma_f32_16x16x4_f32(false, ones2, false, b1,  (short)0, dA, false, false);
        dT = __builtin_amdgcn_wmma_f32_16x16x4_f32(false, ones2, false, tb1, (short)0, dT, false, false);
    }
#else
    v16h ones16;
    for (int j = 0; j < 16; ++j) ones16[j] = (_Float16)1.0f;
    for (int it = 0; it < 4; ++it) {
        const int idx = base + it * 512 + lane * 16;
        v16h bo, bt;
        for (int j = 0; j < 16; ++j) {
            const float o  = rowO[idx + j];
            const int   tv = rowT[idx + j];
            bo[j] = (_Float16)o;
            bt[j] = (_Float16)(o * (float)tv);
        }
        dA = __builtin_amdgcn_wmma_f32_16x16x32_f16(false, ones16, false, bo, (short)0, dA, false, false);
        dT = __builtin_amdgcn_wmma_f32_16x16x32_f16(false, ones16, false, bt, (short)0, dT, false, false);
    }
#endif
    float a = 0.f, t = 0.f;
    for (int v = 0; v < 8; ++v) { a += dA[v]; t += dT[v]; }
    for (int off = 16; off; off >>= 1) {
        a += __shfl_down(a, off, 32);
        t += __shfl_down(t, off, 32);
    }
    __shared__ float pa[4], pt[4];
    if (lane == 0) { pa[wave] = a * (1.0f / 16.0f); pt[wave] = t * (1.0f / 16.0f); }
    __syncthreads();
    if (threadIdx.x == 0) {
        sAll[row] = pa[0] + pa[1] + pa[2] + pa[3];
        sTgt[row] = pt[0] + pt[1] + pt[2] + pt[3];
    }
}

// ---------------------------------------------------------------------------
// Kernel 4: closed-form 3-level softmax per (s,b) + gather correction.
// Single block -> deterministic reduction; writes out[0] = num / den.
// ---------------------------------------------------------------------------
__global__ __launch_bounds__(256) void ocd_finalize(const float* __restrict__ outputs,
                                                    const int* __restrict__ targets,
                                                    const float* __restrict__ sAll,
                                                    const float* __restrict__ sTgt,
                                                    const float* __restrict__ cs_a,
                                                    const float* __restrict__ q0_a,
                                                    const float* __restrict__ nA_a,
                                                    const float* __restrict__ mask_a,
                                                    const int* __restrict__ cCols,
                                                    const int* __restrict__ cSteps,
                                                    const int* __restrict__ cCnt,
                                                    float* __restrict__ out) {
    float num = 0.f, den = 0.f;
    for (int idx = threadIdx.x; idx < S_DIM * B_DIM; idx += 256) {
        const int b = idx % B_DIM;
        const int s = idx / B_DIM;
        const float m = mask_a[idx];
        den += m;
        if (m == 0.f) continue;
        const float* __restrict__ rowO = outputs + (size_t)idx * L_DIM;
        const int cnt = cCnt[b];
        float gather = 0.f;
        bool cons0 = false;
        for (int j = 0; j < cnt; ++j) {
            if (cSteps[b * S_DIM + j] <= s) {
                const int col = cCols[b * S_DIM + j];
                gather += rowO[col];
                if (col == 0) cons0 = true;
            }
        }
        const float out0 = rowO[0];
        const int tgt0 = targets[(size_t)b * L_DIM];
        // sum of outputs over active columns, excluding l==0
        const float sAct = sTgt[idx] - gather - ((tgt0 && !cons0) ? out0 : 0.f);
        const float nA    = nA_a[idx];
        const float nRest = (float)(L_DIM - 1) - nA;
        const float vA = cs_a[idx];      // active value
        const float v0 = vA - 1.f;       // everything else (l != 0)
        const float vE = q0_a[idx];      // exact EOS value
        const float M  = fmaxf(vA, vE);  // vA >= v0 always
        const float eA = __expf((vA - M) / TEMP_);
        const float e0 = __expf((v0 - M) / TEMP_);
        const float eE = __expf((vE - M) / TEMP_);
        const float Z  = nA * eA + nRest * e0 + eE;
        const float pA = eA / Z, p0 = e0 / Z, pE = eE / Z;
        const float plogp = nA * pA * __logf(pA + EPS_)
                          + nRest * p0 * __logf(p0 + EPS_)
                          + pE * __logf(pE + EPS_);
        const float cross = pA * sAct + p0 * (sAll[idx] - sAct - out0) + pE * out0;
        num += (plogp - cross) * (1.0f / (float)L_DIM) * m;
    }
    __shared__ float shN[256], shD[256];
    shN[threadIdx.x] = num;
    shD[threadIdx.x] = den;
    __syncthreads();
    for (int st = 128; st; st >>= 1) {
        if (threadIdx.x < st) {
            shN[threadIdx.x] += shN[threadIdx.x + st];
            shD[threadIdx.x] += shD[threadIdx.x + st];
        }
        __syncthreads();
    }
    if (threadIdx.x == 0) out[0] = shN[0] / shD[0];
}

// ---------------------------------------------------------------------------
extern "C" void kernel_launch(void* const* d_in, const int* in_sizes, int n_in,
                              void* d_out, int out_size, void* d_ws, size_t ws_size,
                              hipStream_t stream) {
    const float* outputs = (const float*)d_in[0];
    const int*   symbols = (const int*)d_in[1];
    const int*   targets = (const int*)d_in[2];
    float* out = (float*)d_out;

    const int SB = S_DIM * B_DIM;
    char* ws = (char*)d_ws;
    float* sAll  = (float*)ws;  ws += (size_t)SB * 4;
    float* sTgt  = (float*)ws;  ws += (size_t)SB * 4;
    float* cs_a  = (float*)ws;  ws += (size_t)SB * 4;
    float* q0_a  = (float*)ws;  ws += (size_t)SB * 4;
    float* nA_a  = (float*)ws;  ws += (size_t)SB * 4;
    float* mk_a  = (float*)ws;  ws += (size_t)SB * 4;
    int* nTgt    = (int*)ws;    ws += (size_t)B_DIM * 4;
    int* cCols   = (int*)ws;    ws += (size_t)B_DIM * S_DIM * 4;
    int* cSteps  = (int*)ws;    ws += (size_t)B_DIM * S_DIM * 4;
    int* cCnt    = (int*)ws;    ws += (size_t)B_DIM * 4;

    ocd_tgt_count<<<B_DIM, 256, 0, stream>>>(targets, nTgt);
    ocd_scan<<<1, 128, 0, stream>>>(symbols, targets, nTgt,
                                    cs_a, q0_a, nA_a, mk_a, cCols, cSteps, cCnt);
    ocd_row_reduce<<<SB, 128, 0, stream>>>(outputs, targets, sAll, sTgt);
    ocd_finalize<<<1, 256, 0, stream>>>(outputs, targets, sAll, sTgt,
                                        cs_a, q0_a, nA_a, mk_a,
                                        cCols, cSteps, cCnt, out);
}
